// MultiHeadSelfAttention_89996744720722
// MI455X (gfx1250) — matrix operand order repaired and checked by reading, not compiled, not run on hardware
//
#include <hip/hip_runtime.h>
#include <hip/hip_bf16.h>

typedef __attribute__((ext_vector_type(16))) _Float16 v16h;
typedef __attribute__((ext_vector_type(8)))  _Float16 v8h;
typedef __attribute__((ext_vector_type(8)))  float    v8f;
typedef __attribute__((ext_vector_type(4)))  unsigned int u32x4;
typedef __attribute__((ext_vector_type(8)))  int          i32x8;
typedef __attribute__((ext_vector_type(4)))  int          i32x4;

#define BATCH 8
#define EMBED 256
#define HEADS 8
#define HDIM  64
#define NTOK  1024

#if __has_builtin(__builtin_amdgcn_tensor_load_to_lds)
#define HAVE_TDM 1
#endif

// ---------------------------------------------------------------------------
// WMMA helper: D = A(16x32 f16) * B(32x16 f16) + C(16x16 f32)
// ---------------------------------------------------------------------------
__device__ __forceinline__ v8f wmma16(v16h a, v16h b, v8f c) {
    return __builtin_amdgcn_wmma_f32_16x16x32_f16(
        /*neg_a=*/false, a, /*neg_b=*/false, b,
        /*c_mod=*/(short)0, c, /*reuse_a=*/false, /*reuse_b=*/false);
}

// Build A fragment from a row pointer with K-chunks at [off + h*16] and
// [off + h*16 + 8] (contiguous k = 16h + s, matching B loads).
__device__ __forceinline__ v16h afrag_row(const _Float16* row, int h) {
    v8h lo = *(const v8h*)(row + h * 16);
    v8h hi = *(const v8h*)(row + h * 16 + 8);
    return __builtin_shufflevector(lo, hi, 0,1,2,3,4,5,6,7,8,9,10,11,12,13,14,15);
}

__device__ __forceinline__ void wait_tensorcnt0() {
#if __has_builtin(__builtin_amdgcn_s_wait_tensorcnt)
    __builtin_amdgcn_s_wait_tensorcnt(0);
#else
    asm volatile("s_wait_tensorcnt 0x0" ::: "memory");
#endif
}

#ifdef HAVE_TDM
// Issue one TDM load of a contiguous [rows x 64 halfs] f16 tile to LDS.
// 2D descriptor: data_size=2B, tensor==tile (no OOB), dim0 stride = 64 elems.
__device__ __forceinline__ void tdm_load_tile_f16(unsigned lds_off,
                                                  unsigned long long gaddr,
                                                  unsigned rows) {
    u32x4 g0;
    g0[0] = 1u;                                            // count=1, user mode
    g0[1] = lds_off;                                       // lds_addr
    g0[2] = (unsigned)(gaddr & 0xFFFFFFFFu);               // global_addr[31:0]
    g0[3] = (unsigned)((gaddr >> 32) & 0x01FFFFFFu)        // global_addr[56:32]
          | (2u << 30);                                    // type=2 ("image")
    i32x8 g1;
    g1[0] = 0x10000;               // workgroup_mask=0, data_size=1 (2 bytes)
    g1[1] = (int)(64u << 16);      // tensor_dim0[15:0]=64 @ bits 63:48
    g1[2] = (int)(rows << 16);     // tensor_dim0 hi=0 ; tensor_dim1[15:0]=rows
    g1[3] = (int)(64u << 16);      // tensor_dim1 hi=0 ; tile_dim0=64
    g1[4] = (int)rows;             // tile_dim1=rows ; tile_dim2=0 (2D)
    g1[5] = 64;                    // tensor_dim0_stride low32 = 64 elems
    g1[6] = 0;                     // stride hi ; tensor_dim1_stride lo
    g1[7] = 0;
    i32x4 z4 = {0, 0, 0, 0};
#if __has_include(<hip/amd_detail/amd_gfx1250_TDM.h>)
    i32x8 z8 = {0, 0, 0, 0, 0, 0, 0, 0};
    __builtin_amdgcn_tensor_load_to_lds(g0, g1, z4, z4, z8, 0);   // clang-23 form
#else
    __builtin_amdgcn_tensor_load_to_lds(g0, g1, z4, z4, 0);       // ROCm 7.2 form
#endif
}
#endif

// ---------------------------------------------------------------------------
// Kernel 1a: (B,C,H,W) fp32 -> token-major (B,N,C) f16
// ---------------------------------------------------------------------------
__global__ void k_pack_x(const float* __restrict__ x, _Float16* __restrict__ X16) {
    int idx = blockIdx.x * 256 + threadIdx.x;      // == ((b*EMBED + c)*NTOK + n)
    int n = idx % NTOK;
    int c = (idx / NTOK) % EMBED;
    int b = idx / (NTOK * EMBED);
    X16[((size_t)(b * NTOK + n)) * EMBED + c] = (_Float16)x[idx];
}

// Kernel 1b: fp32 -> f16 copy (weights)
__global__ void k_cvt(const float* __restrict__ s, _Float16* __restrict__ d, int n) {
    int i = blockIdx.x * 256 + threadIdx.x;
    if (i < n) d[i] = (_Float16)s[i];
}

// ---------------------------------------------------------------------------
// Kernel 2: QKV projection.  Y = X @ W^T + b, split heads, f16 out.
// One wave computes one 16x16 output tile. Q gets *0.125 folded in.
// ---------------------------------------------------------------------------
__global__ __launch_bounds__(256) void k_qkv(
    const _Float16* __restrict__ X16,
    const _Float16* __restrict__ Wq16, const _Float16* __restrict__ Wk16,
    const _Float16* __restrict__ Wv16,
    const float* __restrict__ bq, const float* __restrict__ bk,
    const float* __restrict__ bv,
    _Float16* __restrict__ Qo, _Float16* __restrict__ Ko, _Float16* __restrict__ Vo) {
    int wid  = threadIdx.x >> 5;
    int lane = threadIdx.x & 31;
    int n = lane & 15, h = lane >> 4;

    int t    = blockIdx.x * 8 + wid;
    int proj = t >> 13;
    int rem  = t & 8191;
    int b    = rem >> 10;
    int rem2 = rem & 1023;
    int mt   = rem2 >> 4;
    int nt   = rem2 & 15;

    const _Float16* W16  = (proj == 0) ? Wq16 : ((proj == 1) ? Wk16 : Wv16);
    const float*    bias = (proj == 0) ? bq   : ((proj == 1) ? bk   : bv);
    _Float16*       Out  = (proj == 0) ? Qo   : ((proj == 1) ? Ko   : Vo);

    int d = nt * 16 + n;
    float bval = bias[d];
    v8f acc;
#pragma unroll
    for (int r = 0; r < 8; r++) acc[r] = bval;

    const _Float16* Xrow = X16 + ((size_t)(b * NTOK + mt * 16 + n)) * EMBED;
    const _Float16* Wrow = W16 + (size_t)d * EMBED;

#pragma unroll
    for (int kc = 0; kc < EMBED; kc += 32) {
        v16h a  = afrag_row(Xrow + kc, h);
        v16h bf = *(const v16h*)(Wrow + kc + h * 16);
        acc = wmma16(a, bf, acc);
    }

    float scale = (proj == 0) ? 0.125f : 1.0f;   // 1/sqrt(64) folded into Q
    int head = d >> 6, hd = d & 63;
    _Float16* outp = Out + ((size_t)(b * HEADS + head)) * NTOK * HDIM + hd;
#pragma unroll
    for (int r = 0; r < 8; r++) {
        int q = mt * 16 + h * 8 + r;
        outp[(size_t)q * HDIM] = (_Float16)(acc[r] * scale);
    }
}

// ---------------------------------------------------------------------------
// Kernel 3: flash attention.  8 waves/block; each wave owns 16 query rows of
// one (b,head); stream over 32-key blocks.  K tile staged by the Tensor Data
// Mover (one descriptor per block, issued by wave 0); V tile loaded+transposed
// by all threads, overlapping the DMA.
// ---------------------------------------------------------------------------
__global__ __launch_bounds__(256) void k_attn(
    const _Float16* __restrict__ Qg, const _Float16* __restrict__ Kg,
    const _Float16* __restrict__ Vg, _Float16* __restrict__ Ctx) {
    __shared__ __align__(32) _Float16 ldsK[32 * 64];        // [key][d]
    __shared__ __align__(32) _Float16 ldsVT[64 * 32];       // [d][key]
    __shared__ __align__(32) _Float16 ldsP[8 * 16 * 32];    // per-wave P tile

    int wid  = threadIdx.x >> 5;
    int lane = threadIdx.x & 31;
    int n = lane & 15, h = lane >> 4;

    int qb = blockIdx.x & 7;
    int bh = blockIdx.x >> 3;
    int mt = qb * 8 + wid;

    const _Float16* Qh = Qg + (size_t)bh * NTOK * HDIM;
    const _Float16* Kh = Kg + (size_t)bh * NTOK * HDIM;
    const _Float16* Vh = Vg + (size_t)bh * NTOK * HDIM;

#ifdef HAVE_TDM
    unsigned ldsK_off =
        (unsigned)(__SIZE_TYPE__)(__attribute__((address_space(3))) _Float16*)ldsK;
#endif

    // Resident Q fragments (pre-scaled by 1/8 in projection)
    const _Float16* qrow = Qh + (size_t)(mt * 16 + n) * HDIM;
    v16h qf[2];
#pragma unroll
    for (int g = 0; g < 2; g++) qf[g] = afrag_row(qrow + g * 32, h);

    v8f o[4];
#pragma unroll
    for (int tt = 0; tt < 4; tt++)
#pragma unroll
        for (int r = 0; r < 8; r++) o[tt][r] = 0.0f;
    float rmax[8], rsum[8];
#pragma unroll
    for (int r = 0; r < 8; r++) { rmax[r] = -1e30f; rsum[r] = 0.0f; }

    _Float16* myP = ldsP + wid * (16 * 32);

    for (int kb = 0; kb < NTOK / 32; kb++) {
        __syncthreads();   // previous iteration done with ldsK/ldsVT
#ifdef HAVE_TDM
        if (wid == 0) {    // one TDM descriptor stages the whole K tile
            tdm_load_tile_f16(ldsK_off,
                              (unsigned long long)(Kh + (size_t)kb * 32 * 64), 32u);
        }
#else
        {
            const v8h* src = (const v8h*)(Kh + (size_t)kb * 32 * 64);
            ((v8h*)ldsK)[threadIdx.x] = src[threadIdx.x];
        }
#endif
        {
            // V block, transposed into ldsVT[d][key] (overlaps the K DMA)
            int key = threadIdx.x >> 3;
            int d0  = (threadIdx.x & 7) * 8;
            v8h vv = *(const v8h*)(Vh + (size_t)(kb * 32 + key) * 64 + d0);
#pragma unroll
            for (int j = 0; j < 8; j++) ldsVT[(d0 + j) * 32 + key] = vv[j];
            if (kb + 1 < NTOK / 32)   // global_prefetch_b8 of next V block
                __builtin_prefetch(Vh + (size_t)(kb + 1) * 32 * 64 + threadIdx.x * 8, 0, 1);
        }
#ifdef HAVE_TDM
        if (wid == 0) wait_tensorcnt0();
#endif
        __syncthreads();

        // S tiles: keys [kb*32, kb*32+16) and [+16, +32)
        v8f s0, s1;
#pragma unroll
        for (int r = 0; r < 8; r++) { s0[r] = 0.0f; s1[r] = 0.0f; }
#pragma unroll
        for (int g = 0; g < 2; g++) {
            v16h b0 = *(const v16h*)(ldsK + n * 64 + g * 32 + h * 16);
            v16h b1 = *(const v16h*)(ldsK + (n + 16) * 64 + g * 32 + h * 16);
            s0 = wmma16(qf[g], b0, s0);
            s1 = wmma16(qf[g], b1, s1);
        }

        // Online softmax: each row lives across the 16 lanes of one half-wave
#pragma unroll
        for (int r = 0; r < 8; r++) {
            float mc = fmaxf(s0[r], s1[r]);
#pragma unroll
            for (int msk = 1; msk < 16; msk <<= 1)
                mc = fmaxf(mc, __shfl_xor(mc, msk, 32));
            float mn    = fmaxf(rmax[r], mc);
            float alpha = __expf(rmax[r] - mn);
            float p0 = __expf(s0[r] - mn);
            float p1 = __expf(s1[r] - mn);
            float sc = p0 + p1;
#pragma unroll
            for (int msk = 1; msk < 16; msk <<= 1)
                sc += __shfl_xor(sc, msk, 32);
            rsum[r] = rsum[r] * alpha + sc;
            rmax[r] = mn;
            o[0][r] *= alpha; o[1][r] *= alpha; o[2][r] *= alpha; o[3][r] *= alpha;
            myP[(r + h * 8) * 32 + n]      = (_Float16)p0;
            myP[(r + h * 8) * 32 + 16 + n] = (_Float16)p1;
        }

        // ctx += P(16x32) x V(32x64)
        v16h pf = afrag_row(myP + n * 32, h);
#pragma unroll
        for (int tt = 0; tt < 4; tt++) {
            v16h bf = *(const v16h*)(ldsVT + (tt * 16 + n) * 32 + h * 16);
            o[tt] = wmma16(pf, bf, o[tt]);
        }
    }

    // Epilogue: normalize and store ctx as tokens (B,N,C) f16
    int b = bh >> 3, head = bh & 7;
#pragma unroll
    for (int r = 0; r < 8; r++) {
        float inv = 1.0f / rsum[r];
        int q = mt * 16 + h * 8 + r;
        _Float16* dst = Ctx + ((size_t)(b * NTOK + q)) * EMBED + head * HDIM;
#pragma unroll
        for (int tt = 0; tt < 4; tt++)
            dst[tt * 16 + n] = (_Float16)(o[tt][r] * inv);
    }
}

// ---------------------------------------------------------------------------
// Kernel 4: output projection.  out(B,C,H,W) = Ctx @ Wo^T + bo  (fp32 out)
// ---------------------------------------------------------------------------
__global__ __launch_bounds__(256) void k_out(
    const _Float16* __restrict__ C16, const _Float16* __restrict__ Wo16,
    const float* __restrict__ bo, float* __restrict__ out) {
    int wid  = threadIdx.x >> 5;
    int lane = threadIdx.x & 31;
    int n = lane & 15, h = lane >> 4;

    int t   = blockIdx.x * 8 + wid;
    int b   = t >> 10;
    int rem = t & 1023;
    int mt  = rem >> 4;
    int nt  = rem & 15;

    int d = nt * 16 + n;
    float bval = bo[d];
    v8f acc;
#pragma unroll
    for (int r = 0; r < 8; r++) acc[r] = bval;

    const _Float16* Xrow = C16 + ((size_t)(b * NTOK + mt * 16 + n)) * EMBED;
    const _Float16* Wrow = Wo16 + (size_t)d * EMBED;
#pragma unroll
    for (int kc = 0; kc < EMBED; kc += 32) {
        v16h a  = afrag_row(Xrow + kc, h);
        v16h bf = *(const v16h*)(Wrow + kc + h * 16);
        acc = wmma16(a, bf, acc);
    }

    float* orow = out + ((size_t)(b * EMBED + d)) * NTOK + mt * 16 + h * 8;
    *(float4*)(orow)     = make_float4(acc[0], acc[1], acc[2], acc[3]);
    *(float4*)(orow + 4) = make_float4(acc[4], acc[5], acc[6], acc[7]);
}

// ---------------------------------------------------------------------------
// Host-side launch
// ---------------------------------------------------------------------------
extern "C" void kernel_launch(void* const* d_in, const int* in_sizes, int n_in,
                              void* d_out, int out_size, void* d_ws, size_t ws_size,
                              hipStream_t stream) {
    const float* x  = (const float*)d_in[0];
    const float* Wq = (const float*)d_in[1];
    const float* bq = (const float*)d_in[2];
    const float* Wk = (const float*)d_in[3];
    const float* bk = (const float*)d_in[4];
    const float* Wv = (const float*)d_in[5];
    const float* bv = (const float*)d_in[6];
    const float* Wo = (const float*)d_in[7];
    const float* bo = (const float*)d_in[8];
    float* out = (float*)d_out;

    _Float16* ws = (_Float16*)d_ws;
    const size_t nX   = (size_t)BATCH * NTOK * EMBED;
    const size_t nW   = (size_t)EMBED * EMBED;
    const size_t nQKV = (size_t)BATCH * HEADS * NTOK * HDIM;
    _Float16* X16  = ws;
    _Float16* Wq16 = X16 + nX;
    _Float16* Wk16 = Wq16 + nW;
    _Float16* Wv16 = Wk16 + nW;
    _Float16* Wo16 = Wv16 + nW;
    _Float16* Qb   = Wo16 + nW;
    _Float16* Kb   = Qb + nQKV;
    _Float16* Vb   = Kb + nQKV;
    _Float16* Ctx  = Vb + nQKV;

    k_pack_x<<<(int)(nX / 256), 256, 0, stream>>>(x, X16);
    k_cvt<<<(int)(nW / 256), 256, 0, stream>>>(Wq, Wq16, (int)nW);
    k_cvt<<<(int)(nW / 256), 256, 0, stream>>>(Wk, Wk16, (int)nW);
    k_cvt<<<(int)(nW / 256), 256, 0, stream>>>(Wv, Wv16, (int)nW);
    k_cvt<<<(int)(nW / 256), 256, 0, stream>>>(Wo, Wo16, (int)nW);

    k_qkv<<<3072, 256, 0, stream>>>(X16, Wq16, Wk16, Wv16, bq, bk, bv, Qb, Kb, Vb);
    k_attn<<<512, 256, 0, stream>>>(Qb, Kb, Vb, Ctx);
    k_out<<<1024, 256, 0, stream>>>(Ctx, Wo16, bo, out);
}
